// Real2DMambaBlock_66082366816703
// MI455X (gfx1250) — compile-verified, matrix operand order checked by reading, not run on hardware
//
#include <hip/hip_runtime.h>
#include <hip/hip_bf16.h>

// ---------------- Types for CDNA5 WMMA ----------------
typedef __attribute__((ext_vector_type(16))) __bf16 v16bf;
typedef __attribute__((ext_vector_type(8)))  float  v8f;

union BfTile { uint4 u[2]; v16bf v; };

// Load one WMMA operand tile (A row-major activations or B = weight rows,
// since D = X @ W^T means B[k,n] = W[n,k] i.e. B column n == W row n).
__device__ __forceinline__ void load_tile(BfTile& t, const __bf16* rowBaseAtK0, int hi) {
    const uint4* p = (const uint4*)rowBaseAtK0;
    t.u[0] = p[hi];
    t.u[1] = p[2 + hi];
}

#define WMMA_BF16(ACC, A, B) \
    ACC = __builtin_amdgcn_wmma_f32_16x16x32_bf16(false, (A).v, false, (B).v, (short)0, ACC, false, false)

// ---------------- Problem constants ----------------
#define Bn   2
#define Cc   128
#define Ll   4096          // H*W
#define BL   8192          // B*L
#define DI   256           // d_inner
#define DTR  8             // dt_rank
#define Nst  16            // D_STATE
#define TCH  64            // scan chunk length staged via TDM

// ---------------- Tensor Data Mover (gfx1250) ----------------
#if defined(__has_builtin)
#  if __has_builtin(__builtin_amdgcn_tensor_load_to_lds) && __has_builtin(__builtin_amdgcn_s_wait_tensorcnt)
#    define HAVE_TDM 1
#  endif
#endif
#ifndef HAVE_TDM
#  define HAVE_TDM 0
#endif

#if HAVE_TDM
typedef __attribute__((ext_vector_type(4))) unsigned int tdm_u32x4;
typedef __attribute__((ext_vector_type(8))) int          tdm_i32x8;
typedef __attribute__((ext_vector_type(4))) int          tdm_i32x4;

// Build a <=2D D# descriptor and issue TENSOR_LOAD_TO_LDS.
// data_size_code: 1 = 2 bytes, 2 = 4 bytes (elements are in data_size units).
// tile1 == 0 -> 1-D tile. For 2D: tensor_dim1 = tile1 rows, stride0 = row stride.
__device__ __forceinline__ void tdm_copy(unsigned lds_off, const void* gsrc,
                                         unsigned data_size_code,
                                         unsigned tile0, unsigned tile1,
                                         unsigned dim0, unsigned stride0) {
    unsigned long long ga = (unsigned long long)gsrc;
    tdm_u32x4 g0;
    g0[0] = 1u;                                                 // count=1 (valid user D#)
    g0[1] = lds_off;                                            // lds_addr (bytes)
    g0[2] = (unsigned)ga;                                       // global_addr[31:0]
    g0[3] = (unsigned)((ga >> 32) & 0x01FFFFFFu) | 0x80000000u; // global_addr[56:32] | type=2
    tdm_i32x8 g1;
    g1[0] = (int)(data_size_code << 16);                        // wg_mask=0, data_size
    g1[1] = (int)((dim0 & 0xFFFFu) << 16);                      // tensor_dim0[15:0] @ bits63:48
    g1[2] = (int)(((dim0 >> 16) & 0xFFFFu) |                    // tensor_dim0[31:16]
                  ((tile1 & 0xFFFFu) << 16));                   // tensor_dim1[15:0]
    g1[3] = (int)((tile0 & 0xFFFFu) << 16);                     // tensor_dim1[31:16]=0 | tile_dim0
    g1[4] = (int)(tile1 & 0xFFFFu);                             // tile_dim1 (0 = 1-D), tile_dim2=0
    g1[5] = (int)stride0;                                       // tensor_dim0_stride[31:0]
    g1[6] = 0;                                                  // stride0 hi, dim1_stride lo
    g1[7] = 0;
    tdm_i32x4 z4 = {0, 0, 0, 0};
#if __clang_major__ >= 23
    tdm_i32x8 z8 = {0, 0, 0, 0, 0, 0, 0, 0};
    __builtin_amdgcn_tensor_load_to_lds(g0, g1, z4, z4, z8, 0);
#else
    __builtin_amdgcn_tensor_load_to_lds(g0, g1, z4, z4, 0);
#endif
}
#endif // HAVE_TDM

// ============================================================
// Weight prep
// ============================================================
__global__ void k_cvt_win(const float* __restrict__ W_in, __bf16* __restrict__ w) {
    int id = blockIdx.x * blockDim.x + threadIdx.x;   // 512*128
    if (id < 2 * DI * Cc) w[id] = (__bf16)W_in[id];
}

// W_eff[o][d*256+c] = sum_j W_fuse[o, d*128+j] * W_out[j, c]
__global__ void k_weff(const float* __restrict__ W_fuse, const float* __restrict__ W_out,
                       __bf16* __restrict__ weff) {
    int id = blockIdx.x * blockDim.x + threadIdx.x;   // 128*1024
    if (id >= Cc * 4 * DI) return;
    int o   = id >> 10;
    int col = id & 1023;
    int d = col >> 8, c = col & 255;
    float acc = 0.f;
    #pragma unroll 4
    for (int j = 0; j < Cc; ++j)
        acc += W_fuse[o * (4 * Cc) + d * Cc + j] * W_out[j * DI + c];
    weff[(size_t)o * 1024 + col] = (__bf16)acc;
}

// ============================================================
// LayerNorm over C; emits row-major (ord 0) and column-major (ord 1) orderings.
// ============================================================
__global__ void k_ln(const float* __restrict__ x, const float* __restrict__ ln_w,
                     const float* __restrict__ ln_b, __bf16* __restrict__ xn) {
    int row = blockIdx.x;          // b*L + l   (8192 rows)
    int c   = threadIdx.x;         // 0..127
    int b = row >> 12, l = row & (Ll - 1);
    float v = x[((size_t)(b * Cc + c)) * Ll + l];
    float s = v, s2 = v * v;
    #pragma unroll
    for (int m = 16; m; m >>= 1) { s += __shfl_xor(s, m, 32); s2 += __shfl_xor(s2, m, 32); }
    __shared__ float ls[4], ls2[4];
    if ((c & 31) == 0) { ls[c >> 5] = s; ls2[c >> 5] = s2; }
    __syncthreads();
    s  = ls[0] + ls[1] + ls[2] + ls[3];
    s2 = ls2[0] + ls2[1] + ls2[2] + ls2[3];
    float mu  = s * (1.f / Cc);
    float var = s2 * (1.f / Cc) - mu * mu;
    float r   = rsqrtf(var + 1e-5f);
    float y   = (v - mu) * r * ln_w[c] + ln_b[c];
    int lcm = (l & 63) * 64 + (l >> 6);
    xn[((size_t)(b * Ll) + l) * Cc + c]      = (__bf16)y;
    xn[((size_t)BL + b * Ll + lcm) * Cc + c] = (__bf16)y;
}

// ============================================================
// xz = xn @ W_in^T : M=16384, N=512, K=128   (WMMA bf16)
// ============================================================
__global__ void k_gemm_in(const __bf16* __restrict__ xn, const __bf16* __restrict__ w,
                          __bf16* __restrict__ xz) {
    int m0   = blockIdx.x * 16;
    int wave = threadIdx.x >> 5;
    int lane = threadIdx.x & 31;
    int hi = lane >> 4, l16 = lane & 15;
    int n0 = wave * 64;
    v8f a0 = {}, a1 = {}, a2 = {}, a3 = {};
    const __bf16* arow = xn + (size_t)(m0 + l16) * Cc;
    #pragma unroll
    for (int kk = 0; kk < 4; ++kk) {
        int k0 = kk * 32;
        BfTile a, b0, b1, b2, b3;
        load_tile(a,  arow + k0, hi);
        load_tile(b0, w + (size_t)(n0 +  0 + l16) * Cc + k0, hi);
        load_tile(b1, w + (size_t)(n0 + 16 + l16) * Cc + k0, hi);
        load_tile(b2, w + (size_t)(n0 + 32 + l16) * Cc + k0, hi);
        load_tile(b3, w + (size_t)(n0 + 48 + l16) * Cc + k0, hi);
        WMMA_BF16(a0, a, b0);
        WMMA_BF16(a1, a, b1);
        WMMA_BF16(a2, a, b2);
        WMMA_BF16(a3, a, b3);
    }
    #pragma unroll
    for (int vv = 0; vv < 8; ++vv) {
        int m = m0 + vv + 8 * hi;
        __bf16* orow = xz + (size_t)m * 512;
        orow[n0 +  0 + l16] = (__bf16)a0[vv];
        orow[n0 + 16 + l16] = (__bf16)a1[vv];
        orow[n0 + 32 + l16] = (__bf16)a2[vv];
        orow[n0 + 48 + l16] = (__bf16)a3[vv];
    }
}

// ============================================================
// Causal depthwise conv (k=4) + bias + SiLU, per direction.
// ============================================================
__global__ void k_conv(const __bf16* __restrict__ xz, const float* __restrict__ conv_w,
                       const float* __restrict__ conv_b, __bf16* __restrict__ xc) {
    int blk = blockIdx.x;              // (d*2+b)*4096 + t
    int c   = threadIdx.x;             // 0..255
    int t  = blk & (Ll - 1);
    int db = blk >> 12;
    int b = db & 1, d = db >> 1;
    int ord = d >> 1, rev = d & 1;
    size_t zrow = (size_t)ord * BL + (size_t)b * Ll;
    float acc = conv_b[c];
    #pragma unroll
    for (int k = 0; k < 4; ++k) {
        int s = t - 3 + k;
        if (s >= 0) {
            int ss = rev ? (Ll - 1 - s) : s;
            acc += conv_w[c * 4 + k] * (float)xz[(zrow + ss) * 512 + c];
        }
    }
    float sil = acc / (1.f + __expf(-acc));
    xc[((size_t)db * Ll + t) * DI + c] = (__bf16)sil;
}

// ============================================================
// x_dbl = xc @ W_xp^T  (40 outputs per (d,b,t); K=256)
// ============================================================
__global__ void k_xdbl(const __bf16* __restrict__ xc, const float* __restrict__ W_xp,
                       float* __restrict__ xdbl) {
    int blk = blockIdx.x;
    int j   = threadIdx.x;             // 64 threads, 40 active
    if (j >= DTR + 2 * Nst) return;
    const __bf16* row = xc + (size_t)blk * DI;
    const float*  wr  = W_xp + j * DI;
    float acc = 0.f;
    #pragma unroll 8
    for (int k = 0; k < DI; ++k) acc += (float)row[k] * wr[k];
    xdbl[(size_t)blk * 40 + j] = acc;
}

// ============================================================
// Selective scan with TDM double-buffered LDS staging.
// Block = 8 waves = 16 channels of one (d,b). Lane = (channel half, state n).
// Per 64-step chunk, wave 0 DMAs: x_dbl rows (2560 dwords, 1-D tile) and the
// block's 16-channel xc slice (16x64 bf16, 2-D tile, row stride 256).
// ============================================================
__global__ void k_scan(const float* __restrict__ xdbl, const __bf16* __restrict__ xc,
                       const __bf16* __restrict__ xz, const float* __restrict__ W_dt,
                       const float* __restrict__ b_dt, const float* __restrict__ A_log,
                       const float* __restrict__ Dp, __bf16* __restrict__ yall) {
    __shared__ float  s_xd[2][TCH * 40];
    __shared__ __bf16 s_xh[2][TCH * 16];

    int tid  = threadIdx.x;
    int wave = tid >> 5;
    int lane = tid & 31;
    int n    = lane & 15;
    int half = lane >> 4;
    int d      = blockIdx.x >> 5;              // 32 blocks per direction
    int b      = (blockIdx.x >> 4) & 1;
    int chbase = (blockIdx.x & 15) * 16;       // 16 contiguous channels per block
    int ch     = chbase + 2 * wave + half;
    int ord = d >> 1, rev = d & 1;

    float A_val = -__expf(A_log[ch * Nst + n]);
    float wdt[DTR];
    #pragma unroll
    for (int r = 0; r < DTR; ++r) wdt[r] = W_dt[ch * DTR + r];
    float bdt = b_dt[ch];
    float dpv = Dp[ch];

    const size_t baseT = (size_t)(d * Bn + b) * Ll;
    const size_t zbase = (size_t)ord * BL + (size_t)b * Ll;

#if HAVE_TDM
    unsigned lds_xd0 = (unsigned)(unsigned long long)&s_xd[0][0];
    unsigned lds_xd1 = (unsigned)(unsigned long long)&s_xd[1][0];
    unsigned lds_xh0 = (unsigned)(unsigned long long)&s_xh[0][0];
    unsigned lds_xh1 = (unsigned)(unsigned long long)&s_xh[1][0];
    if (wave == 0) {
        tdm_copy(lds_xd0, xdbl + baseT * 40,          2, TCH * 40, 0,  TCH * 40, TCH * 40);
        tdm_copy(lds_xh0, xc + baseT * DI + chbase,   1, 16,       TCH, 16,      DI);
    }
#endif

    const int NCHUNK = Ll / TCH;
    float h = 0.f;
    for (int c = 0; c < NCHUNK; ++c) {
        int cur = c & 1;
        int t0  = c * TCH;
#if HAVE_TDM
        if (wave == 0) {
            if (c + 1 < NCHUNK) {
                size_t t1 = (size_t)(c + 1) * TCH;
                unsigned dst_xd = (cur ? lds_xd0 : lds_xd1);
                unsigned dst_xh = (cur ? lds_xh0 : lds_xh1);
                tdm_copy(dst_xd, xdbl + (baseT + t1) * 40,        2, TCH * 40, 0,  TCH * 40, TCH * 40);
                tdm_copy(dst_xh, xc + (baseT + t1) * DI + chbase, 1, 16,       TCH, 16,      DI);
                __builtin_amdgcn_s_wait_tensorcnt(2);  // current chunk complete
            } else {
                __builtin_amdgcn_s_wait_tensorcnt(0);
            }
        }
        __syncthreads();
#else
        for (int i = tid; i < TCH * 40; i += 256)
            s_xd[cur][i] = xdbl[(baseT + t0) * 40 + i];
        for (int i = tid; i < TCH * 16; i += 256)
            s_xh[cur][i] = xc[(baseT + t0 + (i >> 4)) * DI + chbase + (i & 15)];
        __syncthreads();
#endif
        const float*  xd0 = &s_xd[cur][0];
        const __bf16* xh0 = &s_xh[cur][0];
        for (int tl = 0; tl < TCH; ++tl) {
            const float* xd = xd0 + tl * 40;
            float dta = bdt;
            #pragma unroll
            for (int r = 0; r < DTR; ++r) dta += xd[r] * wdt[r];
            float dtv = (dta > 20.f) ? dta : log1pf(__expf(dta));   // softplus
            float Bv  = xd[DTR + n];
            float Cv  = xd[DTR + Nst + n];
            float xh  = (float)xh0[tl * 16 + 2 * wave + half];
            float dA  = __expf(dtv * A_val);
            h = dA * h + dtv * Bv * xh;
            float p = h * Cv;
            p += __shfl_xor(p, 1, 32);
            p += __shfl_xor(p, 2, 32);
            p += __shfl_xor(p, 4, 32);
            p += __shfl_xor(p, 8, 32);
            if (n == 0) {
                int t = t0 + tl;
                int s = rev ? (Ll - 1 - t) : t;
                float zv = (float)xz[(zbase + s) * 512 + DI + ch];
                float y  = p + xh * dpv;
                y *= zv / (1.f + __expf(-zv));                      // * silu(z)
                int pos = (ord == 0) ? s : ((s & 63) * 64 + (s >> 6));
                yall[((size_t)b * Ll + pos) * 1024 + d * DI + ch] = (__bf16)y;
            }
        }
        __syncthreads();
    }
}

// ============================================================
// out = x + (yall @ W_eff^T + b_fuse), transposed to (B,C,H,W).
// M=8192, N=128, K=1024 (WMMA bf16).
// ============================================================
__global__ void k_fuse(const __bf16* __restrict__ yall, const __bf16* __restrict__ weff,
                       const float* __restrict__ x, const float* __restrict__ b_fuse,
                       float* __restrict__ out) {
    int m0   = blockIdx.x * 16;
    int wave = threadIdx.x >> 5;       // 0..1
    int lane = threadIdx.x & 31;
    int hi = lane >> 4, l16 = lane & 15;
    int n0 = wave * 64;
    v8f a0 = {}, a1 = {}, a2 = {}, a3 = {};
    const __bf16* arow = yall + (size_t)(m0 + l16) * 1024;
    for (int kk = 0; kk < 32; ++kk) {
        int k0 = kk * 32;
        BfTile a, b0, b1, b2, b3;
        load_tile(a,  arow + k0, hi);
        load_tile(b0, weff + (size_t)(n0 +  0 + l16) * 1024 + k0, hi);
        load_tile(b1, weff + (size_t)(n0 + 16 + l16) * 1024 + k0, hi);
        load_tile(b2, weff + (size_t)(n0 + 32 + l16) * 1024 + k0, hi);
        load_tile(b3, weff + (size_t)(n0 + 48 + l16) * 1024 + k0, hi);
        WMMA_BF16(a0, a, b0);
        WMMA_BF16(a1, a, b1);
        WMMA_BF16(a2, a, b2);
        WMMA_BF16(a3, a, b3);
    }
    #pragma unroll
    for (int vv = 0; vv < 8; ++vv) {
        int m = m0 + vv + 8 * hi;
        int b = m >> 12, p = m & (Ll - 1);
        float vals[4] = { a0[vv], a1[vv], a2[vv], a3[vv] };
        #pragma unroll
        for (int j = 0; j < 4; ++j) {
            int cc = n0 + j * 16 + l16;
            size_t oi = ((size_t)(b * Cc + cc)) * Ll + p;
            out[oi] = x[oi] + vals[j] + b_fuse[cc];
        }
    }
}

// ============================================================
// Host launcher
// ============================================================
extern "C" void kernel_launch(void* const* d_in, const int* in_sizes, int n_in,
                              void* d_out, int out_size, void* d_ws, size_t ws_size,
                              hipStream_t stream) {
    const float* x      = (const float*)d_in[0];
    const float* ln_w   = (const float*)d_in[1];
    const float* ln_b   = (const float*)d_in[2];
    const float* W_in   = (const float*)d_in[3];
    const float* conv_w = (const float*)d_in[4];
    const float* conv_b = (const float*)d_in[5];
    const float* W_xp   = (const float*)d_in[6];
    const float* W_dt   = (const float*)d_in[7];
    const float* b_dt   = (const float*)d_in[8];
    const float* A_log  = (const float*)d_in[9];
    const float* Dp     = (const float*)d_in[10];
    const float* W_out  = (const float*)d_in[11];
    const float* W_fuse = (const float*)d_in[12];
    const float* b_fuse = (const float*)d_in[13];
    float* out = (float*)d_out;

    char* wp = (char*)d_ws;
    auto take = [&](size_t bytes) -> void* {
        void* r = (void*)wp;
        wp += (bytes + 255) & ~(size_t)255;
        return r;
    };
    __bf16* w_in_bf = (__bf16*)take((size_t)2 * DI * Cc * 2);
    __bf16* weff    = (__bf16*)take((size_t)Cc * 4 * DI * 2);
    __bf16* xn      = (__bf16*)take((size_t)2 * BL * Cc * 2);
    __bf16* xz      = (__bf16*)take((size_t)2 * BL * 512 * 2);
    __bf16* xc      = (__bf16*)take((size_t)4 * BL * DI * 2);
    float*  xdbl    = (float*) take((size_t)4 * BL * 40 * 4);
    __bf16* yall    = (__bf16*)take((size_t)BL * 1024 * 2);

    k_cvt_win<<<(2 * DI * Cc + 255) / 256, 256, 0, stream>>>(W_in, w_in_bf);
    k_weff   <<<(Cc * 4 * DI + 255) / 256, 256, 0, stream>>>(W_fuse, W_out, weff);
    k_ln     <<<BL, Cc, 0, stream>>>(x, ln_w, ln_b, xn);
    k_gemm_in<<<(2 * BL) / 16, 256, 0, stream>>>(xn, w_in_bf, xz);
    k_conv   <<<4 * Bn * Ll, DI, 0, stream>>>(xz, conv_w, conv_b, xc);
    k_xdbl   <<<4 * Bn * Ll, 64, 0, stream>>>(xc, W_xp, xdbl);
    k_scan   <<<128, 256, 0, stream>>>(xdbl, xc, xz, W_dt, b_dt, A_log, Dp, yall);
    k_fuse   <<<BL / 16, 64, 0, stream>>>(yall, weff, x, b_fuse, out);
    (void)in_sizes; (void)n_in; (void)out_size; (void)ws_size;
}
